// NetworkBasic_45054206935257
// MI455X (gfx1250) — matrix operand order repaired and checked by reading, not compiled, not run on hardware
//
#include <hip/hip_runtime.h>
#include <cmath>

typedef __attribute__((ext_vector_type(16))) _Float16 v16h;
typedef __attribute__((ext_vector_type(8)))  _Float16 v8h;
typedef __attribute__((ext_vector_type(8)))  float    v8f;

#define DEVINL __device__ __forceinline__

// ---------------------------------------------------------------------------
// IIR scan: y = x (*) (t/tau)e^{1-t/tau}  ==  P[t]=x[t]+aP, Q[t]=a(Q+P), y=cQ
// One thread per (b,c,h,w) site, T=64 contiguous along last dim, float4 steps.
// Wave working set = 8KB contiguous -> lives in WGP$ across the 16 iterations.
// ---------------------------------------------------------------------------
__global__ void psp_iir_kernel(const float* __restrict__ x, float* __restrict__ y,
                               int nsites, float a, float c)
{
    int s = blockIdx.x * blockDim.x + threadIdx.x;
    if (s >= nsites) return;
    const float4* xr = (const float4*)(x + (size_t)s * 64);
    float4*       yr = (float4*)(y + (size_t)s * 64);
    float P = 0.f, Q = 0.f;
#pragma unroll
    for (int i = 0; i < 16; ++i) {
        if (i + 1 < 16) __builtin_prefetch(&xr[i + 1], 0, 3);  // global_prefetch_b8
        float4 v = xr[i];
        float4 o;
        Q = a * (Q + P); P = fmaf(a, P, v.x); o.x = c * Q;
        Q = a * (Q + P); P = fmaf(a, P, v.y); o.y = c * Q;
        Q = a * (Q + P); P = fmaf(a, P, v.z); o.z = c * Q;
        Q = a * (Q + P); P = fmaf(a, P, v.w); o.w = c * Q;
        yr[i] = o;
    }
}

// ---------------------------------------------------------------------------
// Fused spike_layer (refractory as 2-state IIR, c_r = -2*theta*e/tauRef)
// + optional psp(s, tauSr) of the produced spike train. In-place capable.
// ---------------------------------------------------------------------------
__global__ void spike_psp_kernel(const float* __restrict__ u, float* __restrict__ out,
                                 int nsites, float theta, float ar, float cr,
                                 float as, float cs, int do_psp)
{
    int s = blockIdx.x * blockDim.x + threadIdx.x;
    if (s >= nsites) return;
    const float4* ur = (const float4*)(u + (size_t)s * 64);
    float4*       orow = (float4*)(out + (size_t)s * 64);
    float P = 0.f, Q = 0.f;    // refractory IIR state (driven by spikes)
    float Ps = 0.f, Qs = 0.f;  // next-layer PSP IIR state
#pragma unroll
    for (int i = 0; i < 16; ++i) {
        if (i + 1 < 16) __builtin_prefetch(&ur[i + 1], 0, 3);
        float4 uv = ur[i];
        float4 o;
#define SNN_STEP(comp)                                                        \
        {                                                                     \
            Q = ar * (Q + P);                                                 \
            float v  = uv.comp + cr * Q;                                      \
            float sp = (v >= theta) ? 1.f : 0.f;                              \
            P = fmaf(ar, P, sp);                                              \
            if (do_psp) { Qs = as * (Qs + Ps); Ps = fmaf(as, Ps, sp);         \
                          o.comp = cs * Qs; }                                 \
            else        { o.comp = sp; }                                      \
        }
        SNN_STEP(x) SNN_STEP(y) SNN_STEP(z) SNN_STEP(w)
#undef SNN_STEP
        orow[i] = o;
    }
}

// ---------------------------------------------------------------------------
// Helpers for WMMA kernels
// ---------------------------------------------------------------------------
DEVINL int iclamp(int v, int lo, int hi) { v = v < lo ? lo : v; return v > hi ? hi : v; }

// Build the A fragment from a zero-padded 16xK f16 weight tile in LDS:
// lane m = lane%16, half hl = lane/16; rows are two aligned 16B chunks.
DEVINL v16h load_a_frag(const _Float16* __restrict__ swp, int Kpad, int m, int hl, int k0)
{
    const _Float16* base = swp + m * Kpad + k0;
    v8h lo = *(const v8h*)(base + hl * 16);           // K = k0 + hl*16 + [0..7]
    v8h hi = *(const v8h*)(base + hl * 16 + 8);       // K = k0 + hl*16 + 8 + [0..7]
    return __builtin_shufflevector(lo, hi, 0, 1, 2, 3, 4, 5, 6, 7,
                                   8, 9, 10, 11, 12, 13, 14, 15);
}

// ---------------------------------------------------------------------------
// conv1: 5x5, Cin=2 -> Cout=8, pad=2, per pixel. GEMM: M=8(->16),
// K=50(->64, two 16x16x32 f16 WMMAs), N=16 consecutive timesteps.
// All bounds logic is half-uniform; loads are clamped + selected (no branches)
// and coalesce along t (16 lanes x 4B = 64B per element).
// ---------------------------------------------------------------------------
__global__ void conv1_wmma_kernel(const float* __restrict__ psp1,
                                  const float* __restrict__ w1,
                                  float* __restrict__ u1)
{
    __shared__ _Float16 swp[16 * 64];                 // zero-padded 16x64 weight tile
    for (int i = threadIdx.x; i < 16 * 64; i += blockDim.x) {
        int m = i >> 6, k = i & 63;
        swp[i] = (_Float16)((m < 8 && k < 50) ? w1[m * 50 + k] : 0.f);
    }
    __syncthreads();

    int wave = (blockIdx.x * blockDim.x + threadIdx.x) >> 5;
    int lane = threadIdx.x & 31;
    int tt  = wave & 3;                // t-chunk
    int pix = (wave >> 2) & 4095;      // pixel in 64x64
    int b   = wave >> 14;
    int y = pix >> 6, x = pix & 63;
    int n = lane & 15, hl = lane >> 4;
    int t = (tt << 4) + n;
    int m = n;                         // A-matrix row index (semantic)

    v8f acc = {};
#pragma unroll
    for (int j = 0; j < 2; ++j) {
        int k0 = 32 * j;
        v16h afr = load_a_frag(swp, 64, m, hl, k0);
        v16h bfr;
#pragma unroll
        for (int e = 0; e < 16; ++e) {
            int k = k0 + hl * 16 + e;                 // half-uniform
            int kk = k > 49 ? 49 : k;
            int ci = kk / 25, rem = kk - ci * 25;
            int ky = rem / 5, kx = rem - ky * 5;
            int yy = y + ky - 2, xk = x + kx - 2;
            bool valid = (k < 50) & (yy >= 0) & (yy < 64) & (xk >= 0) & (xk < 64);
            int yc = iclamp(yy, 0, 63), xc = iclamp(xk, 0, 63);
            float v = psp1[((((b * 2 + ci) * 64 + yc) * 64 + xc) << 6) + t];
            bfr[e] = (_Float16)(valid ? v : 0.f);
        }
        acc = __builtin_amdgcn_wmma_f32_16x16x32_f16(false, afr, false, bfr,
                                                     (short)0, acc, false, false);
    }
    if (hl == 0) {                      // rows m=0..7 in VGPR r of lanes 0..15
        int base = ((b * 8 * 64 + y) * 64 + x) * 64 + t;
#pragma unroll
        for (int r = 0; r < 8; ++r)
            u1[base + r * (64 * 64 * 64)] = acc[r];   // 16 lanes -> 64B along t
    }
}

// ---------------------------------------------------------------------------
// deconv2x: ConvTranspose2d(k=2,s=2), Cin=8 -> Cout=2. Non-overlapping:
// GEMM rows = (o,i,j) (8 valid), K=8(->32), one WMMA, N=16 timesteps.
// ---------------------------------------------------------------------------
__global__ void deconv_wmma_kernel(const float* __restrict__ psp2,
                                   const float* __restrict__ w2,
                                   float* __restrict__ u2)
{
    __shared__ _Float16 swp[16 * 32];
    for (int i = threadIdx.x; i < 16 * 32; i += blockDim.x) {
        int m = i >> 5, k = i & 31;
        // w2 flat (c,o,i,j): index = c*8 + (o*4+i*2+j) = k*8 + m
        swp[i] = (_Float16)((m < 8 && k < 8) ? w2[k * 8 + m] : 0.f);
    }
    __syncthreads();

    int wave = (blockIdx.x * blockDim.x + threadIdx.x) >> 5;
    int lane = threadIdx.x & 31;
    int tt  = wave & 3;
    int pix = (wave >> 2) & 4095;
    int b   = wave >> 14;
    int y = pix >> 6, x = pix & 63;
    int n = lane & 15, hl = lane >> 4;
    int t = (tt << 4) + n;
    int m = n;

    v16h afr = load_a_frag(swp, 32, m, hl, 0);
    v16h bfr;
#pragma unroll
    for (int e = 0; e < 16; ++e) {
        int k = hl * 16 + e;
        int kc = k > 7 ? 7 : k;
        float v = psp2[((((b * 8 + kc) * 64 + y) * 64 + x) << 6) + t];
        bfr[e] = (_Float16)((k < 8) ? v : 0.f);
    }
    v8f acc = {};
    acc = __builtin_amdgcn_wmma_f32_16x16x32_f16(false, afr, false, bfr,
                                                 (short)0, acc, false, false);
    if (hl == 0) {
#pragma unroll
        for (int r = 0; r < 8; ++r) {
            int o = r >> 2, ii = (r >> 1) & 1, jj = r & 1;
            int Y = 2 * y + ii, X = 2 * x + jj;
            u2[((((b * 2 + o) * 128 + Y) * 128 + X) << 6) + t] = acc[r];
        }
    }
}

// ---------------------------------------------------------------------------
// conv3: 3x3, Cin=2 -> Cout=2, pad=1, on 128x128 grid, K=18(->32), one WMMA,
// N=16 timesteps, plus fused bilinear-2x skip from psp1 (half-pixel weights
// 0.75/0.25, edge clamped). Skip coords/weights are wave-uniform scalars.
// ---------------------------------------------------------------------------
__global__ void conv3_wmma_kernel(const float* __restrict__ psp3,
                                  const float* __restrict__ psp1,
                                  const float* __restrict__ w3,
                                  float* __restrict__ u3)
{
    __shared__ _Float16 swp[16 * 32];
    for (int i = threadIdx.x; i < 16 * 32; i += blockDim.x) {
        int m = i >> 5, k = i & 31;
        swp[i] = (_Float16)((m < 2 && k < 18) ? w3[m * 18 + k] : 0.f);
    }
    __syncthreads();

    int wave = (blockIdx.x * blockDim.x + threadIdx.x) >> 5;
    int lane = threadIdx.x & 31;
    int tt  = wave & 3;
    int pix = (wave >> 2) & 16383;     // pixel in 128x128
    int b   = wave >> 16;
    int y = pix >> 7, x = pix & 127;
    int n = lane & 15, hl = lane >> 4;
    int t = (tt << 4) + n;
    int m = n;

    v16h afr = load_a_frag(swp, 32, m, hl, 0);
    v16h bfr;
#pragma unroll
    for (int e = 0; e < 16; ++e) {
        int k = hl * 16 + e;
        int kk = k > 17 ? 17 : k;
        int ci = kk / 9, rem = kk - ci * 9;
        int ky = rem / 3, kx = rem - ky * 3;
        int yy = y + ky - 1, xk = x + kx - 1;
        bool valid = (k < 18) & (yy >= 0) & (yy < 128) & (xk >= 0) & (xk < 128);
        int yc = iclamp(yy, 0, 127), xc = iclamp(xk, 0, 127);
        float v = psp3[((((b * 2 + ci) * 128 + yc) * 128 + xc) << 6) + t];
        bfr[e] = (_Float16)(valid ? v : 0.f);
    }
    v8f acc = {};
    acc = __builtin_amdgcn_wmma_f32_16x16x32_f16(false, afr, false, bfr,
                                                 (short)0, acc, false, false);
    if (hl == 0) {
        // Wave-uniform bilinear-2x setup on the 64x64 source grid.
        int y0, y1, x0, x1; float wy0, wy1, wx0, wx1;
        if (y & 1) { y0 = y >> 1; y1 = y0 + 1; wy0 = 0.75f; wy1 = 0.25f; }
        else       { y1 = y >> 1; y0 = y1 - 1; wy0 = 0.25f; wy1 = 0.75f; }
        if (x & 1) { x0 = x >> 1; x1 = x0 + 1; wx0 = 0.75f; wx1 = 0.25f; }
        else       { x1 = x >> 1; x0 = x1 - 1; wx0 = 0.25f; wx1 = 0.75f; }
        y0 = iclamp(y0, 0, 63); y1 = iclamp(y1, 0, 63);
        x0 = iclamp(x0, 0, 63); x1 = iclamp(x1, 0, 63);
#pragma unroll
        for (int r = 0; r < 2; ++r) {
            int cb = (b * 2 + r) * 64;
            float s00 = psp1[(((cb + y0) * 64 + x0) << 6) + t];
            float s01 = psp1[(((cb + y0) * 64 + x1) << 6) + t];
            float s10 = psp1[(((cb + y1) * 64 + x0) << 6) + t];
            float s11 = psp1[(((cb + y1) * 64 + x1) << 6) + t];
            float skip = wy0 * (wx0 * s00 + wx1 * s01) + wy1 * (wx0 * s10 + wx1 * s11);
            u3[((((b * 2 + r) * 128 + y) * 128 + x) << 6) + t] = acc[r] + skip;
        }
    }
}

// ---------------------------------------------------------------------------
// Orchestration. Workspace: P1 (psp1, 16 MiB) @ 0; U1 (64 MiB) @ 16 MiB;
// U2 (64 MiB) @ 80 MiB. conv3 writes into d_out; spike3 finishes in place.
// ---------------------------------------------------------------------------
extern "C" void kernel_launch(void* const* d_in, const int* in_sizes, int n_in,
                              void* d_out, int out_size, void* d_ws, size_t ws_size,
                              hipStream_t stream)
{
    (void)in_sizes; (void)n_in; (void)out_size; (void)ws_size;
    const float* spikeInput = (const float*)d_in[0];
    const float* w1 = (const float*)d_in[1];
    const float* w2 = (const float*)d_in[2];
    const float* w3 = (const float*)d_in[3];
    float* out = (float*)d_out;
    char*  ws  = (char*)d_ws;
    float* P1 = (float*)ws;                               // (8,2,64,64,64)
    float* U1 = (float*)(ws + (size_t)16 * 1024 * 1024);  // (8,8,64,64,64)
    float* U2 = (float*)(ws + (size_t)80 * 1024 * 1024);  // (8,2,128,128,64)

    const float E = 2.718281828459045f;
    auto decay = [](float tau) { return expf(-1.0f / tau); };

    // psp1 = psp(spikeInput, tau=1)
    psp_iir_kernel<<<(8 * 2 * 64 * 64) / 256, 256, 0, stream>>>(
        spikeInput, P1, 8 * 2 * 64 * 64, decay(1.f), E / 1.f);

    // u1 = conv5x5(psp1, w1) via WMMA  (131072 waves, 8 per block)
    conv1_wmma_kernel<<<16384, 256, 0, stream>>>(P1, w1, U1);

    // s1 = spike(u1, theta=30, tauRef=1); U1 <- psp(s1, tau=2)   (fused, in place)
    spike_psp_kernel<<<(8 * 8 * 64 * 64) / 256, 256, 0, stream>>>(
        U1, U1, 8 * 8 * 64 * 64,
        30.f, decay(1.f), -2.f * 30.f * E / 1.f, decay(2.f), E / 2.f, 1);

    // u2 = deconv2x(psp2, w2) via WMMA
    deconv_wmma_kernel<<<16384, 256, 0, stream>>>(U1, w2, U2);

    // s2 = spike(u2, theta=50, tauRef=2); U2 <- psp(s2, tau=4)   (fused, in place)
    spike_psp_kernel<<<(8 * 2 * 128 * 128) / 256, 256, 0, stream>>>(
        U2, U2, 8 * 2 * 128 * 128,
        50.f, decay(2.f), -2.f * 50.f * E / 2.f, decay(4.f), E / 4.f, 1);

    // u3 = conv3x3(psp3, w3) + bilinear2x(psp1)  -> d_out (WMMA + fused skip)
    conv3_wmma_kernel<<<65536, 256, 0, stream>>>(U2, P1, w3, out);

    // s3 = spike(u3, theta=100, tauRef=4)  in place on d_out
    spike_psp_kernel<<<(8 * 2 * 128 * 128) / 256, 256, 0, stream>>>(
        out, out, 8 * 2 * 128 * 128,
        100.f, decay(4.f), -2.f * 100.f * E / 4.f, 0.f, 0.f, 0);
}
